// PureConstantModel_70205535420732
// MI455X (gfx1250) — compile-verified, matrix operand order checked by reading, not run on hardware
//
#include <hip/hip_runtime.h>
#include <hip/hip_bf16.h>
#include <math.h>

// ---------------- model dims ----------------
#define BB   32
#define SS   512
#define DD   384
#define HH   12
#define HD   32
#define LL   6
#define FF   1536
#define TT   (BB * SS)     // 16384 tokens

typedef __bf16 bf16x16 __attribute__((ext_vector_type(16)));
typedef __bf16 bf16x8  __attribute__((ext_vector_type(8)));
typedef float  f32x8   __attribute__((ext_vector_type(8)));

__device__ __forceinline__ float sigmf(float x) { return 1.0f / (1.0f + __expf(-x)); }
__device__ __forceinline__ float geluf(float x) {
    float x3 = x * x * x;
    return 0.5f * x * (1.0f + tanhf(0.7978845608028654f * (x + 0.044715f * x3)));
}
__device__ __forceinline__ float wave_sum(float v) {
#pragma unroll
    for (int off = 16; off > 0; off >>= 1) v += __shfl_xor(v, off, 32);
    return v;
}

// ---------------- f32 -> bf16 conversion ----------------
__global__ __launch_bounds__(256) void convert_bf16_kernel(
    const float* __restrict__ in, __bf16* __restrict__ out, int n)
{
    int i = blockIdx.x * 256 + threadIdx.x;
    if (i < n) out[i] = (__bf16)in[i];
}

// f32 [K,N] -> bf16 [N,K] (transposed weights so GEMM B-tiles are K-contiguous)
__global__ __launch_bounds__(256) void transpose_convert_kernel(
    const float* __restrict__ in, __bf16* __restrict__ out, int K, int N)
{
    size_t i = (size_t)blockIdx.x * 256 + threadIdx.x;
    if (i >= (size_t)K * N) return;
    int n = (int)(i % N), k = (int)(i / N);
    out[(size_t)n * K + k] = (__bf16)in[i];
}

// ---------------- embeddings + layernorm ----------------
__global__ __launch_bounds__(128) void embed_ln_kernel(
    const int* __restrict__ ids, const float* __restrict__ wemb,
    const float* __restrict__ pemb, const float* __restrict__ temb,
    const float* __restrict__ g, const float* __restrict__ b,
    float* __restrict__ x)
{
    const int t = blockIdx.x;
    const int s = t % SS;
    const int id = ids[t];
    const int tid = threadIdx.x;
    __shared__ float red[128];

    float val[3];
    float ps = 0.f, ps2 = 0.f;
#pragma unroll
    for (int e = 0; e < 3; ++e) {
        int d = tid + 128 * e;
        float ev = wemb[(size_t)id * DD + d] + pemb[(size_t)s * DD + d] + temb[d];
        val[e] = ev; ps += ev; ps2 += ev * ev;
    }
    red[tid] = ps; __syncthreads();
    for (int st = 64; st > 0; st >>= 1) { if (tid < st) red[tid] += red[tid + st]; __syncthreads(); }
    float mean = red[0] / (float)DD; __syncthreads();
    red[tid] = ps2; __syncthreads();
    for (int st = 64; st > 0; st >>= 1) { if (tid < st) red[tid] += red[tid + st]; __syncthreads(); }
    float var = red[0] / (float)DD - mean * mean;
    float inv = rsqrtf(var + 1e-12f);
#pragma unroll
    for (int e = 0; e < 3; ++e) {
        int d = tid + 128 * e;
        x[(size_t)t * DD + d] = (val[e] - mean) * inv * g[d] + b[d];
    }
}

// ---------------- residual add + layernorm ----------------
__global__ __launch_bounds__(128) void add_ln_kernel(
    const float* __restrict__ r, const float* __restrict__ y,
    const float* __restrict__ g, const float* __restrict__ b,
    float* __restrict__ out)
{
    const int t = blockIdx.x;
    const int tid = threadIdx.x;
    __shared__ float red[128];

    float val[3];
    float ps = 0.f, ps2 = 0.f;
#pragma unroll
    for (int e = 0; e < 3; ++e) {
        int d = tid + 128 * e;
        float ev = r[(size_t)t * DD + d] + y[(size_t)t * DD + d];
        val[e] = ev; ps += ev; ps2 += ev * ev;
    }
    red[tid] = ps; __syncthreads();
    for (int st = 64; st > 0; st >>= 1) { if (tid < st) red[tid] += red[tid + st]; __syncthreads(); }
    float mean = red[0] / (float)DD; __syncthreads();
    red[tid] = ps2; __syncthreads();
    for (int st = 64; st > 0; st >>= 1) { if (tid < st) red[tid] += red[tid + st]; __syncthreads(); }
    float var = red[0] / (float)DD - mean * mean;
    float inv = rsqrtf(var + 1e-12f);
#pragma unroll
    for (int e = 0; e < 3; ++e) {
        int d = tid + 128 * e;
        out[(size_t)t * DD + d] = (val[e] - mean) * inv * g[d] + b[d];
    }
}

// ================= bf16 WMMA GEMM with async-LDS double buffering ============
// A [M,K] bf16 row-major, Wt [N,K] bf16 row-major (pre-transposed weights),
// bias [N] f32.  C = act(A @ W + bias), stored f32 or bf16.
// M%128==0, N%128==0, K%32==0.
#define BM 128
#define BN 128
#define BK 32
#define KPAD (BK + 8)        // 80-byte row stride (16B-aligned)
#define ACT_NONE 0
#define ACT_GELU 1

union FragU { bf16x16 v; bf16x8 h[2]; };

// stage a 128x32 bf16 tile (rows of gbase, K-contiguous) into LDS via
// GLOBAL_LOAD_ASYNC_TO_LDS_B128 (ASYNCcnt-tracked, no VGPR data path)
__device__ __forceinline__ void async_stage_tile(
    const __bf16* __restrict__ gbase, int ldg, int row0, int k0,
    __bf16 (*lds)[KPAD], int tid)
{
#pragma unroll
    for (int c = 0; c < 2; ++c) {
        int chunk = tid + c * 256;                 // 512 chunks = 128 rows x 4 segs
        int rr = chunk >> 2, seg = chunk & 3;
        unsigned long long ga = (unsigned long long)(size_t)
            (gbase + (size_t)(row0 + rr) * ldg + k0 + seg * 8);
        unsigned la = (unsigned)(size_t)(&lds[rr][seg * 8]);
        asm volatile("global_load_async_to_lds_b128 %0, %1, off"
                     :: "v"(la), "v"(ga) : "memory");
    }
}

__global__ __launch_bounds__(256) void gemm_bf16_kernel(
    const __bf16* __restrict__ A, const __bf16* __restrict__ Wt,
    const float* __restrict__ bias, void* __restrict__ Cout,
    int M, int N, int K, int act, int store_bf16)
{
    __shared__ __align__(16) __bf16 As[2][BM][KPAD];
    __shared__ __align__(16) __bf16 Bs[2][BN][KPAD];

    const int tid  = threadIdx.x;
    const int wave = tid >> 5, lane = tid & 31;
    const int wm   = wave >> 1;          // 0..3 : 32-row group
    const int wn   = wave & 1;           // 0..1 : 64-col group
    const int hl   = lane >> 4, l16 = lane & 15;
    const int tileM = blockIdx.y * BM, tileN = blockIdx.x * BN;

    f32x8 acc[2][4];
#pragma unroll
    for (int i = 0; i < 2; ++i)
#pragma unroll
        for (int j = 0; j < 4; ++j)
#pragma unroll
            for (int e = 0; e < 8; ++e) acc[i][j][e] = 0.0f;

    int buf = 0;
    async_stage_tile(A,  K, tileM, 0, As[0], tid);
    async_stage_tile(Wt, K, tileN, 0, Bs[0], tid);
    asm volatile("s_wait_asynccnt 0x0" ::: "memory");
    __syncthreads();

    for (int k0 = 0; k0 < K; k0 += BK) {
        if (k0 + BK < K) {              // prefetch next tiles into other buffer
            async_stage_tile(A,  K, tileM, k0 + BK, As[buf ^ 1], tid);
            async_stage_tile(Wt, K, tileN, k0 + BK, Bs[buf ^ 1], tid);
        }
        // fragments: per-lane K-halves are contiguous -> two ds_load_b128 each
        FragU af[2], bfr[4];
#pragma unroll
        for (int i = 0; i < 2; ++i) {
            const __bf16* rp = &As[buf][wm * 32 + i * 16 + l16][0];
            af[i].h[0] = *(const bf16x8*)(rp + hl * 8);
            af[i].h[1] = *(const bf16x8*)(rp + 16 + hl * 8);
        }
#pragma unroll
        for (int j = 0; j < 4; ++j) {
            const __bf16* rp = &Bs[buf][wn * 64 + j * 16 + l16][0];
            bfr[j].h[0] = *(const bf16x8*)(rp + hl * 8);
            bfr[j].h[1] = *(const bf16x8*)(rp + 16 + hl * 8);
        }
#pragma unroll
        for (int i = 0; i < 2; ++i)
#pragma unroll
            for (int j = 0; j < 4; ++j)
                acc[i][j] = __builtin_amdgcn_wmma_f32_16x16x32_bf16(
                    false, af[i].v, false, bfr[j].v, (short)0, acc[i][j], false, false);

        asm volatile("s_wait_asynccnt 0x0" ::: "memory");
        __syncthreads();
        buf ^= 1;
    }

    // epilogue: D layout -> VGPR r: m = 16*i + 8*hl + r, n = l16
#pragma unroll
    for (int i = 0; i < 2; ++i)
#pragma unroll
        for (int j = 0; j < 4; ++j) {
            int n = tileN + wn * 64 + j * 16 + l16;
            float bv = bias[n];
#pragma unroll
            for (int rr = 0; rr < 8; ++rr) {
                int m = tileM + wm * 32 + i * 16 + hl * 8 + rr;
                float vv = acc[i][j][rr] + bv;
                if (act == ACT_GELU) vv = geluf(vv);
                if (store_bf16) ((__bf16*)Cout)[(size_t)m * N + n] = (__bf16)vv;
                else            ((float*)Cout)[(size_t)m * N + n] = vv;
            }
        }
}

// ---------------- q/k L2 normalization (wave per contiguous 32-vector) -------
__global__ __launch_bounds__(256) void qk_norm_kernel(float* __restrict__ q, float* __restrict__ k)
{
    int w = (blockIdx.x * 256 + threadIdx.x) >> 5;
    int lane = threadIdx.x & 31;
    if (w >= TT * HH) return;
    size_t idx = (size_t)w * HD + lane;
    float qv = q[idx], kv = k[idx];
    float qn = sqrtf(wave_sum(qv * qv));
    float kn = sqrtf(wave_sum(kv * kv));
    q[idx] = qv / (qn + 1e-6f);
    k[idx] = kv / (kn + 1e-6f);
}

// ---------------- beta = sigmoid(x @ Wb) * mask  (wave per (t,h)) ------------
__global__ __launch_bounds__(256) void beta_kernel(
    const float* __restrict__ x, const float* __restrict__ Wb,
    const int* __restrict__ am, float* __restrict__ beta)
{
    int w = (blockIdx.x * 256 + threadIdx.x) >> 5;
    int lane = threadIdx.x & 31;
    if (w >= TT * HH) return;
    int t = w / HH, h = w % HH;
    float s = 0.f;
#pragma unroll
    for (int i = 0; i < DD / 32; ++i) {
        int d = lane + 32 * i;
        s += x[(size_t)t * DD + d] * Wb[(size_t)d * HH + h];
    }
    s = wave_sum(s);
    if (lane == 0)
        beta[(size_t)t * HH + h] = sigmf(s) * (float)am[t];
}

// ---------------- hierarchical delta-rule scan (wave per (b,h)) --------------
__global__ __launch_bounds__(256) void deltanet_scan_kernel(
    const float* __restrict__ q, const float* __restrict__ k,
    const float* __restrict__ v, const float* __restrict__ beta,
    const float* __restrict__ dfast, const float* __restrict__ dslow,
    const float* __restrict__ mixl, float* __restrict__ o)
{
    int w = (blockIdx.x * 256 + threadIdx.x) >> 5;   // 0 .. B*H-1
    int lane = threadIdx.x & 31;                     // = v-index
    if (w >= BB * HH) return;
    int b = w / HH, h = w % HH;

    const float af = sigmf(dfast[h]);
    const float as = sigmf(dslow[h]);
    const float g  = sigmf(mixl[h]);

    float Sf[HD], Ss[HD];                            // lane holds column v; index = k
#pragma unroll
    for (int i = 0; i < HD; ++i) { Sf[i] = 0.f; Ss[i] = 0.f; }

    for (int t = 0; t < SS; ++t) {
        size_t idx = ((size_t)(b * SS + t) * DD) + h * HD + lane;
        float qt = q[idx], kt = k[idx], vt = v[idx];
        float bt = beta[(size_t)(b * SS + t) * HH + h];

        float pf = 0.f, ps = 0.f;
#pragma unroll
        for (int kk = 0; kk < HD; ++kk) {
            float kb = __shfl(kt, kk, 32);
            pf += Sf[kk] * kb;
            ps += Ss[kk] * kb;
        }
        float df = vt - pf;
        float dsv = vt - ps;

        float of = 0.f, os = 0.f;
#pragma unroll
        for (int kk = 0; kk < HD; ++kk) {
            float kb = __shfl(kt, kk, 32);
            float qb = __shfl(qt, kk, 32);
            Sf[kk] = af * Sf[kk] + bt * kb * df;
            Ss[kk] = as * Ss[kk] + bt * kb * dsv;
            of += Sf[kk] * qb;
            os += Ss[kk] * qb;
        }
        o[idx] = g * of + (1.f - g) * os;
    }
}

// ---------------- masked mean pool + L2 normalize ----------------------------
__global__ __launch_bounds__(128) void pool_norm_kernel(
    const float* __restrict__ x, const int* __restrict__ am, float* __restrict__ out)
{
    const int b = blockIdx.x;
    const int tid = threadIdx.x;
    __shared__ float red[128];

    float mc = 0.f;
    for (int s = tid; s < SS; s += 128) mc += (float)am[b * SS + s];
    red[tid] = mc; __syncthreads();
    for (int st = 64; st > 0; st >>= 1) { if (tid < st) red[tid] += red[tid + st]; __syncthreads(); }
    float denom = fmaxf(red[0], 1e-9f);
    __syncthreads();

    float p[3]; float ps2 = 0.f;
#pragma unroll
    for (int e = 0; e < 3; ++e) {
        int d = tid + 128 * e;
        float s = 0.f;
        for (int ss = 0; ss < SS; ++ss)
            s += x[((size_t)(b * SS + ss)) * DD + d] * (float)am[b * SS + ss];
        p[e] = s / denom;
        ps2 += p[e] * p[e];
    }
    red[tid] = ps2; __syncthreads();
    for (int st = 64; st > 0; st >>= 1) { if (tid < st) red[tid] += red[tid + st]; __syncthreads(); }
    float inv = rsqrtf(red[0]);
#pragma unroll
    for (int e = 0; e < 3; ++e) {
        int d = tid + 128 * e;
        out[(size_t)b * DD + d] = p[e] * inv;
    }
}

// ============================ host launch ====================================
extern "C" void kernel_launch(void* const* d_in, const int* in_sizes, int n_in,
                              void* d_out, int out_size, void* d_ws, size_t ws_size,
                              hipStream_t stream)
{
    const int*   ids    = (const int*)  d_in[0];
    const int*   am     = (const int*)  d_in[1];
    const float* wemb   = (const float*)d_in[2];
    const float* pemb   = (const float*)d_in[3];
    const float* temb   = (const float*)d_in[4];
    const float* eg     = (const float*)d_in[5];
    const float* ebias  = (const float*)d_in[6];
    const float* Wq     = (const float*)d_in[7];
    const float* bq     = (const float*)d_in[8];
    const float* Wk     = (const float*)d_in[9];
    const float* bk     = (const float*)d_in[10];
    const float* Wv     = (const float*)d_in[11];
    const float* bv     = (const float*)d_in[12];
    const float* Wb     = (const float*)d_in[13];
    const float* dfast  = (const float*)d_in[14];
    const float* dslow  = (const float*)d_in[15];
    const float* mixl   = (const float*)d_in[16];
    const float* Wo     = (const float*)d_in[17];
    const float* bo     = (const float*)d_in[18];
    const float* alng   = (const float*)d_in[19];
    const float* alnb   = (const float*)d_in[20];
    const float* W1     = (const float*)d_in[21];
    const float* b1     = (const float*)d_in[22];
    const float* W2     = (const float*)d_in[23];
    const float* b2     = (const float*)d_in[24];
    const float* flng   = (const float*)d_in[25];
    const float* flnb   = (const float*)d_in[26];

    // ---------------- workspace layout ----------------
    char* ws = (char*)d_ws;
    size_t cur = 0;
    auto alloc = [&](size_t bytes) { char* p = ws + cur; cur += (bytes + 255) & ~(size_t)255; return p; };

    float*  x    = (float*) alloc((size_t)TT * DD * 4);
    float*  y    = (float*) alloc((size_t)TT * DD * 4);
    float*  q    = (float*) alloc((size_t)TT * DD * 4);
    float*  k    = (float*) alloc((size_t)TT * DD * 4);
    float*  v    = (float*) alloc((size_t)TT * DD * 4);
    float*  o    = (float*) alloc((size_t)TT * DD * 4);
    float*  beta = (float*) alloc((size_t)TT * HH * 4);
    __bf16* xb   = (__bf16*)alloc((size_t)TT * DD * 2);   // also reused for bf16(o)
    __bf16* hb   = (__bf16*)alloc((size_t)TT * FF * 2);   // FFN hidden, written bf16 by GEMM
    __bf16* Wqt  = (__bf16*)alloc((size_t)LL * DD * DD * 2);
    __bf16* Wkt  = (__bf16*)alloc((size_t)LL * DD * DD * 2);
    __bf16* Wvt  = (__bf16*)alloc((size_t)LL * DD * DD * 2);
    __bf16* Wot  = (__bf16*)alloc((size_t)LL * DD * DD * 2);
    __bf16* W1t  = (__bf16*)alloc((size_t)LL * DD * FF * 2);  // [FF][DD] per layer
    __bf16* W2t  = (__bf16*)alloc((size_t)LL * FF * DD * 2);  // [DD][FF] per layer

    const dim3 gemmBlk(256);
    const dim3 gD(DD / BN, TT / BM);    // (3, 128)
    const dim3 gF(FF / BN, TT / BM);    // (12, 128)
    const int cvD  = (TT * DD + 255) / 256;
    const int nVecBlocks = (TT * HH) / 8;
    const int scanBlocks = (BB * HH * 32 + 255) / 256;
    const int gWsmall = (DD * DD + 255) / 256;
    const int gWbig   = (DD * FF + 255) / 256;

    // one-time (per launch) weight transpose+convert to bf16
    for (int l = 0; l < LL; ++l) {
        transpose_convert_kernel<<<gWsmall, 256, 0, stream>>>(Wq + (size_t)l*DD*DD, Wqt + (size_t)l*DD*DD, DD, DD);
        transpose_convert_kernel<<<gWsmall, 256, 0, stream>>>(Wk + (size_t)l*DD*DD, Wkt + (size_t)l*DD*DD, DD, DD);
        transpose_convert_kernel<<<gWsmall, 256, 0, stream>>>(Wv + (size_t)l*DD*DD, Wvt + (size_t)l*DD*DD, DD, DD);
        transpose_convert_kernel<<<gWsmall, 256, 0, stream>>>(Wo + (size_t)l*DD*DD, Wot + (size_t)l*DD*DD, DD, DD);
        transpose_convert_kernel<<<gWbig,   256, 0, stream>>>(W1 + (size_t)l*DD*FF, W1t + (size_t)l*FF*DD, DD, FF);
        transpose_convert_kernel<<<gWbig,   256, 0, stream>>>(W2 + (size_t)l*FF*DD, W2t + (size_t)l*DD*FF, FF, DD);
    }

    embed_ln_kernel<<<TT, 128, 0, stream>>>(ids, wemb, pemb, temb, eg, ebias, x);

    for (int l = 0; l < LL; ++l) {
        convert_bf16_kernel<<<cvD, 256, 0, stream>>>(x, xb, TT * DD);
        gemm_bf16_kernel<<<gD, gemmBlk, 0, stream>>>(xb, Wqt + (size_t)l*DD*DD, bq + l*DD, q, TT, DD, DD, ACT_NONE, 0);
        gemm_bf16_kernel<<<gD, gemmBlk, 0, stream>>>(xb, Wkt + (size_t)l*DD*DD, bk + l*DD, k, TT, DD, DD, ACT_NONE, 0);
        gemm_bf16_kernel<<<gD, gemmBlk, 0, stream>>>(xb, Wvt + (size_t)l*DD*DD, bv + l*DD, v, TT, DD, DD, ACT_NONE, 0);
        beta_kernel<<<nVecBlocks, 256, 0, stream>>>(x, Wb + (size_t)l*DD*HH, am, beta);
        qk_norm_kernel<<<nVecBlocks, 256, 0, stream>>>(q, k);
        deltanet_scan_kernel<<<scanBlocks, 256, 0, stream>>>(
            q, k, v, beta, dfast + l*HH, dslow + l*HH, mixl + l*HH, o);
        convert_bf16_kernel<<<cvD, 256, 0, stream>>>(o, xb, TT * DD);
        gemm_bf16_kernel<<<gD, gemmBlk, 0, stream>>>(xb, Wot + (size_t)l*DD*DD, bo + l*DD, y, TT, DD, DD, ACT_NONE, 0);
        add_ln_kernel<<<TT, 128, 0, stream>>>(x, y, alng + l*DD, alnb + l*DD, x);

        convert_bf16_kernel<<<cvD, 256, 0, stream>>>(x, xb, TT * DD);
        gemm_bf16_kernel<<<gF, gemmBlk, 0, stream>>>(xb, W1t + (size_t)l*FF*DD, b1 + l*FF, hb, TT, FF, DD, ACT_GELU, 1);
        gemm_bf16_kernel<<<gD, gemmBlk, 0, stream>>>(hb, W2t + (size_t)l*DD*FF, b2 + l*DD, y, TT, DD, FF, ACT_NONE, 0);
        add_ln_kernel<<<TT, 128, 0, stream>>>(x, y, flng + l*DD, flnb + l*DD, x);
    }

    pool_norm_kernel<<<BB, 128, 0, stream>>>(x, am, (float*)d_out);
}
